// VQLatentSpace2D_23691039605498
// MI455X (gfx1250) — matrix operand order repaired and checked by reading, not compiled, not run on hardware
//
#include <hip/hip_runtime.h>
#include <hip/hip_bf16.h>
#include <math.h>

#define NUM_EMB 1024
#define EDIM 256
#define NVEC 32768          // 32*32*32 vectors
#define QELEMS 8388608      // 32*256*32*32
#define BROW 264            // 256 + 8 ushort padding (bank-conflict-free LDS stride)

typedef __bf16 v16bf __attribute__((ext_vector_type(16)));
typedef float  v8f   __attribute__((ext_vector_type(8)));
typedef unsigned short u16x8  __attribute__((ext_vector_type(8)));
typedef unsigned short u16x16 __attribute__((ext_vector_type(16)));

__device__ __forceinline__ unsigned short f32_to_bf16_rne(float f) {
  unsigned int u = __float_as_uint(f);
  unsigned int r = u + 0x7FFFu + ((u >> 16) & 1u);
  return (unsigned short)(r >> 16);
}
__device__ __forceinline__ float bf16_bits_to_f32(unsigned short b) {
  return __uint_as_float(((unsigned int)b) << 16);
}

// ---------------------------------------------------------------------------
// Kernel 0: transpose [b,c,h,w] -> X[n=(b,hw), d=c] and split fp32 -> bf16 hi/lo
// ---------------------------------------------------------------------------
__global__ void prep_x_kernel(const float* __restrict__ in,
                              unsigned short* __restrict__ xhi,
                              unsigned short* __restrict__ xlo) {
  __shared__ float tile[32][33];
  int t  = blockIdx.x;
  int b  = t >> 8;            // 32 batches
  int ct = (t >> 5) & 7;      // 8 c-tiles of 32
  int wt = t & 31;            // 32 hw-tiles of 32
  int c0 = ct * 32, hw0 = wt * 32;
  int tx = threadIdx.x & 31;
  int ty = threadIdx.x >> 5;  // 0..7
  const float* src = in + (size_t)b * EDIM * 1024;
#pragma unroll
  for (int i = 0; i < 4; ++i) {
    int row = ty * 4 + i;     // c offset within tile
    tile[row][tx] = src[(size_t)(c0 + row) * 1024 + hw0 + tx];
  }
  __syncthreads();
#pragma unroll
  for (int i = 0; i < 4; ++i) {
    int row = ty * 4 + i;     // hw offset within tile
    float x = tile[tx][row];
    unsigned short hb = f32_to_bf16_rne(x);
    float lof = x - bf16_bits_to_f32(hb);
    unsigned short lb = f32_to_bf16_rne(lof);
    size_t o = (size_t)(b * 1024 + hw0 + row) * EDIM + c0 + tx;
    xhi[o] = hb;
    xlo[o] = lb;
  }
}

// ---------------------------------------------------------------------------
// Kernel 1: codebook bf16 hi/lo split + hnorm[k] = 0.5*||e_k||^2, zero counts
// ---------------------------------------------------------------------------
__global__ void prep_e_kernel(const float* __restrict__ emb,
                              unsigned short* __restrict__ ehi,
                              unsigned short* __restrict__ elo,
                              float* __restrict__ hnorm,
                              int* __restrict__ counts) {
  __shared__ float wsum[8];
  int k = blockIdx.x;
  int d = threadIdx.x;
  float e = emb[(size_t)k * EDIM + d];
  unsigned short hb = f32_to_bf16_rne(e);
  float lof = e - bf16_bits_to_f32(hb);
  ehi[(size_t)k * EDIM + d] = hb;
  elo[(size_t)k * EDIM + d] = f32_to_bf16_rne(lof);
  float s = e * e;
#pragma unroll
  for (int m = 16; m; m >>= 1) s += __shfl_xor(s, m, 32);
  int lane = threadIdx.x & 31, w = threadIdx.x >> 5;
  if (lane == 0) wsum[w] = s;
  __syncthreads();
  if (threadIdx.x == 0) {
    float tot = 0.f;
#pragma unroll
    for (int i = 0; i < 8; ++i) tot += wsum[i];
    hnorm[k] = 0.5f * tot;
    counts[k] = 0;
  }
}

// ---------------------------------------------------------------------------
// Kernel 2: WMMA GEMM (score = x.e - 0.5||e||^2) + per-row argmax -> idx
// One wave = 16 rows x all 1024 codes. bf16 hi/lo split, fp32 accumulate.
// B tiles staged once per block in double-buffered LDS (8x less L2 traffic).
// ---------------------------------------------------------------------------
__global__ void __launch_bounds__(256)
gemm_argmin_kernel(const unsigned short* __restrict__ xhi,
                   const unsigned short* __restrict__ xlo,
                   const unsigned short* __restrict__ ehi,
                   const unsigned short* __restrict__ elo,
                   const float* __restrict__ hnorm,
                   int* __restrict__ idx_out) {
  // [buf][hi/lo][16 rows * BROW]  -> 2*2*16*264*2B = 33 KB
  __shared__ unsigned short Bsh[2][2][16 * BROW];

  int tid  = threadIdx.x;
  int lane = tid & 31;
  int m0   = (blockIdx.x * 8 + (tid >> 5)) * 16;   // row tile base (per wave)
  int col  = lane & 15;
  int hv   = lane >> 4;                            // half-wave select

  // --- A fragments for this wave's 16 rows, all 8 K-chunks, hi and lo ---
  // ISA layout: lanes 0-15 row M=lane, K = k0+{0..7,16..23};
  //             lanes 16-31 row M=lane-16, K = k0+{8..15,24..31}
  v16bf ahi[8], alo[8];
  {
    const unsigned short* xh = xhi + (size_t)(m0 + col) * EDIM;
    const unsigned short* xl = xlo + (size_t)(m0 + col) * EDIM;
#pragma unroll
    for (int ch = 0; ch < 8; ++ch) {
      int base = ch * 32 + hv * 8;
      u16x8 l0 = *(const u16x8*)(xh + base);
      u16x8 h0 = *(const u16x8*)(xh + base + 16);
      u16x16 p0 = __builtin_shufflevector(l0, h0, 0,1,2,3,4,5,6,7,8,9,10,11,12,13,14,15);
      ahi[ch] = __builtin_bit_cast(v16bf, p0);
      u16x8 l1 = *(const u16x8*)(xl + base);
      u16x8 h1 = *(const u16x8*)(xl + base + 16);
      u16x16 p1 = __builtin_shufflevector(l1, h1, 0,1,2,3,4,5,6,7,8,9,10,11,12,13,14,15);
      alo[ch] = __builtin_bit_cast(v16bf, p1);
    }
  }

  // Cooperative B-tile staging: 512 16B-chunks per tile (16 rows x 32 chunks),
  // 256 threads -> 2 chunks each for hi and lo.
  int q0 = tid, q1 = tid + 256;
  int sr0 = q0 >> 5, sc0 = q0 & 31;   // row, chunk-of-8-ushorts
  int sr1 = q1 >> 5, sc1 = q1 & 31;
  uint4 rh0, rh1, rl0, rl1;

  auto stage_load = [&](int ct) {
    const unsigned short* bh = ehi + (size_t)(ct * 16) * EDIM;
    const unsigned short* bl = elo + (size_t)(ct * 16) * EDIM;
    rh0 = *(const uint4*)(bh + sr0 * EDIM + sc0 * 8);
    rh1 = *(const uint4*)(bh + sr1 * EDIM + sc1 * 8);
    rl0 = *(const uint4*)(bl + sr0 * EDIM + sc0 * 8);
    rl1 = *(const uint4*)(bl + sr1 * EDIM + sc1 * 8);
  };
  auto stage_store = [&](int buf) {
    *(uint4*)(&Bsh[buf][0][sr0 * BROW + sc0 * 8]) = rh0;
    *(uint4*)(&Bsh[buf][0][sr1 * BROW + sc1 * 8]) = rh1;
    *(uint4*)(&Bsh[buf][1][sr0 * BROW + sc0 * 8]) = rl0;
    *(uint4*)(&Bsh[buf][1][sr1 * BROW + sc1 * 8]) = rl1;
  };

  float best[8];
  int   bidx[8];
#pragma unroll
  for (int r = 0; r < 8; ++r) { best[r] = -3.0e38f; bidx[r] = 0; }

  stage_load(0);
  stage_store(0);

  for (int ct = 0; ct < 64; ++ct) {
    if (ct < 63) stage_load(ct + 1);    // overlap next tile's global loads
    __syncthreads();                    // buf[ct&1] visible; buf[(ct+1)&1] free
    if (ct < 63) stage_store((ct + 1) & 1);

    int code = ct * 16 + col;
    float hn = hnorm[code];
    v8f c;
#pragma unroll
    for (int r = 0; r < 8; ++r) c[r] = -hn;   // fold -0.5||e||^2 into C init

    // B layout: lane n (col) holds code column's K = hv*8+{0..7} and 16+hv*8+{0..7}
    const unsigned short* bp0 = &Bsh[ct & 1][0][col * BROW + hv * 8];
    const unsigned short* bp1 = &Bsh[ct & 1][1][col * BROW + hv * 8];
#pragma unroll
    for (int ch = 0; ch < 8; ++ch) {
      u16x8 hl = *(const u16x8*)(bp0 + ch * 32);
      u16x8 hh = *(const u16x8*)(bp0 + ch * 32 + 16);
      v16bf bh = __builtin_bit_cast(v16bf,
          __builtin_shufflevector(hl, hh, 0,1,2,3,4,5,6,7,8,9,10,11,12,13,14,15));
      u16x8 ll = *(const u16x8*)(bp1 + ch * 32);
      u16x8 lh = *(const u16x8*)(bp1 + ch * 32 + 16);
      v16bf bl = __builtin_bit_cast(v16bf,
          __builtin_shufflevector(ll, lh, 0,1,2,3,4,5,6,7,8,9,10,11,12,13,14,15));
      c = __builtin_amdgcn_wmma_f32_16x16x32_bf16(false, ahi[ch], false, bh, (short)0, c, false, false);
      c = __builtin_amdgcn_wmma_f32_16x16x32_bf16(false, ahi[ch], false, bl, (short)0, c, false, false);
      c = __builtin_amdgcn_wmma_f32_16x16x32_bf16(false, alo[ch], false, bh, (short)0, c, false, false);
    }
#pragma unroll
    for (int r = 0; r < 8; ++r) {
      if (c[r] > best[r]) { best[r] = c[r]; bidx[r] = code; }
    }
  }

  // cross-lane argmax within each 16-lane column group (lower index wins ties)
#pragma unroll
  for (int r = 0; r < 8; ++r) {
    float v = best[r]; int i = bidx[r];
#pragma unroll
    for (int m = 8; m; m >>= 1) {
      float ov = __shfl_xor(v, m, 32);
      int   oi = __shfl_xor(i, m, 32);
      if (ov > v || (ov == v && oi < i)) { v = ov; i = oi; }
    }
    if (col == 0) idx_out[m0 + hv * 8 + r] = i;  // C layout: M = hv*8 + r
  }
}

// ---------------------------------------------------------------------------
// Kernel 3: histogram of indices (LDS-staged, integer atomics -> deterministic)
// ---------------------------------------------------------------------------
__global__ void hist_kernel(const int* __restrict__ idx, int* __restrict__ counts) {
  __shared__ int lh[NUM_EMB];
  for (int i = threadIdx.x; i < NUM_EMB; i += blockDim.x) lh[i] = 0;
  __syncthreads();
  int n = blockIdx.x * blockDim.x + threadIdx.x;
  atomicAdd(&lh[idx[n]], 1);
  __syncthreads();
  for (int i = threadIdx.x; i < NUM_EMB; i += blockDim.x) {
    int v = lh[i];
    if (v) atomicAdd(&counts[i], v);
  }
}

// ---------------------------------------------------------------------------
// Kernel 4: gather quantized (already [b,c,h,w] layout) + partial SSE per block
// ---------------------------------------------------------------------------
__global__ void quant_loss_kernel(const float* __restrict__ in,
                                  const float* __restrict__ emb,
                                  const int* __restrict__ idx,
                                  float* __restrict__ qout,
                                  float* __restrict__ partial) {
  __shared__ float wsum[8];
  int t  = blockIdx.x * 256 + threadIdx.x;
  int hw = t & 1023;
  int c  = (t >> 10) & 255;
  int b  = t >> 18;
  int n  = b * 1024 + hw;
  float x = in[t];
  int id  = idx[n];
  float e = emb[(size_t)id * EDIM + c];
  qout[t] = e;                      // quantized_st == quantized numerically
  float d = e - x;
  float s = d * d;
#pragma unroll
  for (int m = 16; m; m >>= 1) s += __shfl_xor(s, m, 32);
  int lane = threadIdx.x & 31, w = threadIdx.x >> 5;
  if (lane == 0) wsum[w] = s;
  __syncthreads();
  if (threadIdx.x == 0) {
    float tot = 0.f;
#pragma unroll
    for (int i = 0; i < 8; ++i) tot += wsum[i];
    partial[blockIdx.x] = tot;
  }
}

// ---------------------------------------------------------------------------
// Kernel 5: one-hot encodings fill (128 MB write-once stream -> non-temporal)
// ---------------------------------------------------------------------------
__global__ void enc_fill_kernel(const int* __restrict__ idx, float* __restrict__ enc) {
  size_t t = (size_t)blockIdx.x * 256 + threadIdx.x;
  int n = (int)(t >> 10);
  int k = (int)(t & 1023);
  float v = (idx[n] == k) ? 1.0f : 0.0f;
  __builtin_nontemporal_store(v, enc + t);
}

// ---------------------------------------------------------------------------
// Kernel 6: scalars: loss = 1.25*SSE/(N*D), perplexity = exp(-sum p log(p+eps))
// ---------------------------------------------------------------------------
__global__ void finalize_kernel(const int* __restrict__ counts,
                                const float* __restrict__ partial,
                                float* __restrict__ out_loss,
                                float* __restrict__ out_perp) {
  __shared__ float wsE[32];
  __shared__ float wsS[32];
  int tid = threadIdx.x;  // 1024 threads
  float p = (float)counts[tid] * (1.0f / 32768.0f);
  float ent = p * logf(p + 1e-10f);
  float sse = 0.f;
  for (int i = tid; i < 32768; i += 1024) sse += partial[i];  // fixed order
#pragma unroll
  for (int m = 16; m; m >>= 1) {
    ent += __shfl_xor(ent, m, 32);
    sse += __shfl_xor(sse, m, 32);
  }
  int lane = tid & 31, w = tid >> 5;
  if (lane == 0) { wsE[w] = ent; wsS[w] = sse; }
  __syncthreads();
  if (tid == 0) {
    float te = 0.f, ts = 0.f;
    for (int i = 0; i < 32; ++i) { te += wsE[i]; ts += wsS[i]; }
    *out_perp = expf(-te);
    *out_loss = 1.25f * ts / 8388608.0f;
  }
}

// ---------------------------------------------------------------------------
extern "C" void kernel_launch(void* const* d_in, const int* in_sizes, int n_in,
                              void* d_out, int out_size, void* d_ws, size_t ws_size,
                              hipStream_t stream) {
  (void)in_sizes; (void)n_in; (void)out_size; (void)ws_size;
  const float* in  = (const float*)d_in[0];
  const float* emb = (const float*)d_in[1];

  char* ws = (char*)d_ws;
  unsigned short* xhi = (unsigned short*)(ws + 0);          // 16 MB
  unsigned short* xlo = (unsigned short*)(ws + 16777216);   // 16 MB
  unsigned short* ehi = (unsigned short*)(ws + 33554432);   // 512 KB
  unsigned short* elo = (unsigned short*)(ws + 34078720);   // 512 KB
  float*          hnm = (float*)(ws + 34603008);            // 4 KB
  int*            cnt = (int*)  (ws + 34607104);            // 4 KB
  int*            idx = (int*)  (ws + 34611200);            // 128 KB
  float*          par = (float*)(ws + 34742272);            // 128 KB

  float* out  = (float*)d_out;
  float* qout = out;                  // [32,256,32,32]
  float* loss = out + QELEMS;         // scalar
  float* perp = out + QELEMS + 1;     // scalar
  float* enc  = out + QELEMS + 2;     // [32,32,32,1024]

  prep_x_kernel     <<<8192,   256, 0, stream>>>(in, xhi, xlo);
  prep_e_kernel     <<<1024,   256, 0, stream>>>(emb, ehi, elo, hnm, cnt);
  gemm_argmin_kernel<<<256,    256, 0, stream>>>(xhi, xlo, ehi, elo, hnm, idx);
  hist_kernel       <<<128,    256, 0, stream>>>(idx, cnt);
  quant_loss_kernel <<<32768,  256, 0, stream>>>(in, emb, idx, qout, par);
  enc_fill_kernel   <<<131072, 256, 0, stream>>>(idx, enc);
  finalize_kernel   <<<1,     1024, 0, stream>>>(cnt, par, loss, perp);
}
